// SentimentGRU_57896159150702
// MI455X (gfx1250) — matrix operand order repaired and checked by reading, not compiled, not run on hardware
//
#include <hip/hip_runtime.h>
#include <hip/hip_bf16.h>

// ---------------------------------------------------------------------------
// SentimentGRU on gfx1250 (MI455X):
//   emb gather -> f16 packed into WMMA A-fragment order
//   weights f32 -> f16 packed into WMMA B-fragment order (1KB/fragment,
//   perfectly coalesced global_load_b128 clauses in the scan loop)
//   fused 2-layer GRU scan, WMMA f32_16x16x32_f16, h kept in LDS (f16,
//   row stride padded to 520 elems to kill ds bank conflicts)
//   one workgroup per 16 batch rows, 32 waves each owning 16 hidden units;
//   only intra-WG barriers on the sequential critical path.
// ---------------------------------------------------------------------------

typedef __attribute__((ext_vector_type(16))) _Float16 v16h;
typedef __attribute__((ext_vector_type(8)))  float    v8f;

#define B_   64
#define S_   512
#define E_   256
#define H_   512
#define G3_  (3 * H_)   // 1536
#define HS_  520        // padded LDS row stride (f16 elems), 1040B = 4 banks skew

union FragU {
  uint4 u[2];
  v16h  v;
};

// Opaque zero SGPR: blocks cross-iteration hoist/CSE of t-invariant weight
// loads (previously spilled ~1300 hoisted VGPRs) w/o destroying provenance.
__device__ __forceinline__ unsigned long long opaque_zero() {
  unsigned long long z = 0;
  asm volatile("" : "+s"(z));
  return z;
}

// Packed fragment (A or B already laid out in lane order):
// lane l owns 32 contiguous bytes at block*1KB + l*32.
__device__ __forceinline__ v16h load_frag_packed(const _Float16* base, size_t blk, int lane) {
  const _Float16* p = base + (blk * 32 + (size_t)lane) * 16;
  FragU f;
  f.u[0] = *reinterpret_cast<const uint4*>(p);
  f.u[1] = *reinterpret_cast<const uint4*>(p + 8);
  return f.v;
}

// A-fragment from LDS H (row-major, padded stride HS_):
// lane l -> row r = l&15, kh = l>>4; 16B at k0+kh*8 and k0+16+kh*8.
__device__ __forceinline__ v16h load_a_lds(const _Float16* Hbuf, int k0, int lane) {
  const int r  = lane & 15;
  const int kh = lane >> 4;
  const _Float16* p = Hbuf + r * HS_ + k0 + kh * 8;
  FragU f;
  f.u[0] = *reinterpret_cast<const uint4*>(p);
  f.u[1] = *reinterpret_cast<const uint4*>(p + 16);
  return f.v;
}

__device__ __forceinline__ v8f wmma_f16(v16h a, v16h b, v8f c) {
  return __builtin_amdgcn_wmma_f32_16x16x32_f16(
      /*neg_a=*/false, a, /*neg_b=*/false, b,
      /*c_mod=*/(short)0, c, /*reuse_a=*/false, /*reuse_b=*/false);
}

__device__ __forceinline__ float sigmoidf_(float x) {
  return 1.0f / (1.0f + __expf(-x));
}

// ---------------------------------------------------------------------------
// Prep kernels
// ---------------------------------------------------------------------------

// Embedding gather, written directly in A-fragment packed order.
// dst block (bt*S + t)*8 + kx holds the 16x32 tile rows b = bt*16+(l&15),
// cols e = kx*32 + quarter-pattern(kh, i).
__global__ void embed_pack_kernel(const int* __restrict__ x,
                                  const float* __restrict__ emb,
                                  _Float16* __restrict__ dst, int total) {
  int tid = blockIdx.x * blockDim.x + threadIdx.x;
  if (tid >= total) return;
  int i    = tid & 15;
  int lane = (tid >> 4) & 31;
  int blk  = tid >> 9;
  int r  = lane & 15;
  int kh = lane >> 4;
  int kx = blk & 7;
  int t  = (blk >> 3) & (S_ - 1);
  int bt = blk >> 12;               // /(8*S_)
  int b  = bt * 16 + r;
  int kl = (i < 8) ? (kh * 8 + i) : (16 + kh * 8 + (i - 8));
  int e  = kx * 32 + kl;
  int tok = x[b * S_ + t];
  float v = (tok == 0) ? 0.0f : emb[(size_t)tok * E_ + e];   // PAD_IDX row zeroed
  dst[tid] = (_Float16)v;
}

// Weight pack: rows (n) x cols (k) f32 -> B-fragment order f16.
// Block rt*nk + ks holds cols n = rt*16+(l&15), k = ks*32 + (i<8 ? (l>>4)*8 + i : 16 + (l>>4)*8 + i-8).
__global__ void pack_w_kernel(const float* __restrict__ src,
                              _Float16* __restrict__ dst, int K, int total) {
  int tid = blockIdx.x * blockDim.x + threadIdx.x;
  if (tid >= total) return;
  int i    = tid & 15;
  int lane = (tid >> 4) & 31;
  int blk  = tid >> 9;
  int nk = K >> 5;
  int ks = blk % nk;
  int rt = blk / nk;
  int n  = rt * 16 + (lane & 15);
  int k  = ks * 32 + ((i < 8) ? ((lane >> 4) * 8 + i) : (16 + (lane >> 4) * 8 + (i - 8)));
  dst[tid] = (_Float16)src[(size_t)n * K + k];
}

// ---------------------------------------------------------------------------
// Fused 2-layer GRU scan + FC head.
// grid = 4 (16 batch rows each), block = 1024 (32 waves)
// wave w owns hidden units [16w, 16w+16); gate g row-tile index = g*32 + w.
// ---------------------------------------------------------------------------
__global__ __launch_bounds__(1024) void gru_fused_kernel(
    const _Float16* __restrict__ Xp,    // packed A frags: [(bt*S+t)*8+kx]
    const _Float16* __restrict__ Wih0p, // packed B frags, nk=8
    const _Float16* __restrict__ Whh0p, // packed B frags, nk=16
    const _Float16* __restrict__ Wih1p, // packed B frags, nk=16
    const _Float16* __restrict__ Whh1p, // packed B frags, nk=16
    const float* __restrict__ b_ih0, const float* __restrict__ b_hh0,
    const float* __restrict__ b_ih1, const float* __restrict__ b_hh1,
    const float* __restrict__ W_fc,  const float* __restrict__ b_fc,
    float* __restrict__ out) {
  __shared__ __align__(16) _Float16 H0[16 * HS_];
  __shared__ __align__(16) _Float16 H1[16 * HS_];

  const int tid  = threadIdx.x;
  const int lane = tid & 31;
  const int w    = tid >> 5;       // wave id 0..31
  const int j0   = w * 16;         // hidden-unit tile owned by this wave
  const int bt   = blockIdx.x;
  const int b0   = bt * 16;

  for (int i = tid; i < 16 * HS_; i += 1024) {
    H0[i] = (_Float16)0.0f;
    H1[i] = (_Float16)0.0f;
  }
  __syncthreads();

  // per-lane gate biases (lane l and l+16 share column jl = j0 + (l&15))
  const int jl = j0 + (lane & 15);
  const float bR0  = b_ih0[jl]        + b_hh0[jl];
  const float bZ0  = b_ih0[H_ + jl]   + b_hh0[H_ + jl];
  const float bXN0 = b_ih0[2*H_ + jl];
  const float bHN0 = b_hh0[2*H_ + jl];
  const float bR1  = b_ih1[jl]        + b_hh1[jl];
  const float bZ1  = b_ih1[H_ + jl]   + b_hh1[H_ + jl];
  const float bXN1 = b_ih1[2*H_ + jl];
  const float bHN1 = b_hh1[2*H_ + jl];

  const int kh = lane >> 4;

  for (int t = 0; t < S_; ++t) {
    const unsigned long long z = opaque_zero();
    const _Float16* wih0 = Wih0p + z;
    const _Float16* whh0 = Whh0p + z;
    const _Float16* wih1 = Wih1p + z;
    const _Float16* whh1 = Whh1p + z;

    // ================= layer 0 =================
    // r/z fuse x-part + h-part; n needs xn and hn separate (n = tanh(xn + r*hn))
    v8f aR = {}, aZ = {}, aXN = {}, aHN = {};
    const size_t xblk = ((size_t)bt * S_ + t) * 8;

#pragma unroll
    for (int ks = 0; ks < 8; ++ks) {   // x-part, K = E_ = 256
      v16h a = load_frag_packed(Xp, xblk + ks, lane);
      aR  = wmma_f16(a, load_frag_packed(wih0, (size_t)(     w) * 8 + ks, lane), aR);
      aZ  = wmma_f16(a, load_frag_packed(wih0, (size_t)(32 + w) * 8 + ks, lane), aZ);
      aXN = wmma_f16(a, load_frag_packed(wih0, (size_t)(64 + w) * 8 + ks, lane), aXN);
    }
#pragma unroll
    for (int ks = 0; ks < 16; ++ks) {  // h-part, K = H_ = 512
      v16h a = load_a_lds(H0, ks * 32, lane);
      aR  = wmma_f16(a, load_frag_packed(whh0, (size_t)(     w) * 16 + ks, lane), aR);
      aZ  = wmma_f16(a, load_frag_packed(whh0, (size_t)(32 + w) * 16 + ks, lane), aZ);
      aHN = wmma_f16(a, load_frag_packed(whh0, (size_t)(64 + w) * 16 + ks, lane), aHN);
    }
    __syncthreads();   // all waves done reading H0 for this step

    // combine: D frag element (v, lane) -> m = v + 8*kh, col jl (owner-exclusive)
#pragma unroll
    for (int v = 0; v < 8; ++v) {
      const int m = v + (kh << 3);
      float r  = sigmoidf_(aR[v] + bR0);
      float z0 = sigmoidf_(aZ[v] + bZ0);
      float nn = tanhf(aXN[v] + bXN0 + r * (aHN[v] + bHN0));
      float hp = (float)H0[m * HS_ + jl];
      H0[m * HS_ + jl] = (_Float16)((1.0f - z0) * nn + z0 * hp);
    }
    __syncthreads();   // H0 (= layer-1 input) ready

    // ================= layer 1 =================
    v8f cR = {}, cZ = {}, cXN = {}, cHN = {};
#pragma unroll
    for (int ks = 0; ks < 16; ++ks) {  // x-input = layer-0 output
      v16h a = load_a_lds(H0, ks * 32, lane);
      cR  = wmma_f16(a, load_frag_packed(wih1, (size_t)(     w) * 16 + ks, lane), cR);
      cZ  = wmma_f16(a, load_frag_packed(wih1, (size_t)(32 + w) * 16 + ks, lane), cZ);
      cXN = wmma_f16(a, load_frag_packed(wih1, (size_t)(64 + w) * 16 + ks, lane), cXN);
    }
#pragma unroll
    for (int ks = 0; ks < 16; ++ks) {
      v16h a = load_a_lds(H1, ks * 32, lane);
      cR  = wmma_f16(a, load_frag_packed(whh1, (size_t)(     w) * 16 + ks, lane), cR);
      cZ  = wmma_f16(a, load_frag_packed(whh1, (size_t)(32 + w) * 16 + ks, lane), cZ);
      cHN = wmma_f16(a, load_frag_packed(whh1, (size_t)(64 + w) * 16 + ks, lane), cHN);
    }
    __syncthreads();

#pragma unroll
    for (int v = 0; v < 8; ++v) {
      const int m = v + (kh << 3);
      float r  = sigmoidf_(cR[v] + bR1);
      float z1 = sigmoidf_(cZ[v] + bZ1);
      float nn = tanhf(cXN[v] + bXN1 + r * (cHN[v] + bHN1));
      float hp = (float)H1[m * HS_ + jl];
      H1[m * HS_ + jl] = (_Float16)((1.0f - z1) * nn + z1 * hp);
    }
    __syncthreads();
  }

  // ===== FC head: out[b] = sigmoid(h1[b] . W_fc + b_fc) =====
  if (tid < 16 * 32) {
    const int m = tid >> 5;   // batch row within tile
    float partial = 0.0f;
    for (int j = lane; j < H_; j += 32)
      partial += (float)H1[m * HS_ + j] * W_fc[j];
#pragma unroll
    for (int off = 16; off > 0; off >>= 1)
      partial += __shfl_down(partial, off, 32);
    if (lane == 0)
      out[b0 + m] = sigmoidf_(partial + b_fc[0]);
  }
}

// ---------------------------------------------------------------------------
// launch
// ---------------------------------------------------------------------------
extern "C" void kernel_launch(void* const* d_in, const int* in_sizes, int n_in,
                              void* d_out, int out_size, void* d_ws, size_t ws_size,
                              hipStream_t stream) {
  const int*   x     = (const int*)  d_in[0];
  const float* emb   = (const float*)d_in[1];
  const float* W_ih0 = (const float*)d_in[2];
  const float* W_hh0 = (const float*)d_in[3];
  const float* b_ih0 = (const float*)d_in[4];
  const float* b_hh0 = (const float*)d_in[5];
  const float* W_ih1 = (const float*)d_in[6];
  const float* W_hh1 = (const float*)d_in[7];
  const float* b_ih1 = (const float*)d_in[8];
  const float* b_hh1 = (const float*)d_in[9];
  const float* W_fc  = (const float*)d_in[10];
  const float* b_fc  = (const float*)d_in[11];
  float* out = (float*)d_out;

  // workspace layout (f16 elements)
  const size_t nX    = (size_t)B_ * S_ * E_;   // 8,388,608
  const size_t nWih0 = (size_t)G3_ * E_;       //   393,216
  const size_t nWhh  = (size_t)G3_ * H_;       //   786,432
  _Float16* Xp    = (_Float16*)d_ws;
  _Float16* Wih0p = Xp    + nX;
  _Float16* Whh0p = Wih0p + nWih0;
  _Float16* Wih1p = Whh0p + nWhh;
  _Float16* Whh1p = Wih1p + nWhh;

  embed_pack_kernel<<<(int)((nX + 255) / 256), 256, 0, stream>>>(x, emb, Xp, (int)nX);
  pack_w_kernel<<<(int)((nWih0 + 255) / 256), 256, 0, stream>>>(W_ih0, Wih0p, E_, (int)nWih0);
  pack_w_kernel<<<(int)((nWhh  + 255) / 256), 256, 0, stream>>>(W_hh0, Whh0p, H_, (int)nWhh);
  pack_w_kernel<<<(int)((nWhh  + 255) / 256), 256, 0, stream>>>(W_ih1, Wih1p, H_, (int)nWhh);
  pack_w_kernel<<<(int)((nWhh  + 255) / 256), 256, 0, stream>>>(W_hh1, Whh1p, H_, (int)nWhh);

  gru_fused_kernel<<<B_ / 16, 1024, 0, stream>>>(
      Xp, Wih0p, Whh0p, Wih1p, Whh1p,
      b_ih0, b_hh0, b_ih1, b_hh1, W_fc, b_fc, out);
}
